// MultiHeadSelfAttention_24275155157691
// MI455X (gfx1250) — matrix operand order repaired and checked by reading, not compiled, not run on hardware
//
#include <hip/hip_runtime.h>

// ---------------- problem constants ----------------
constexpr int BATCH = 4;
constexpr int SEQ   = 2048;
constexpr int DM    = 1024;
constexpr int NH    = 16;
constexpr int HD    = 64;             // head dim
constexpr int M_ROWS = BATCH * SEQ;   // 8192

// ---------------- CDNA5 WMMA types ----------------
typedef __attribute__((ext_vector_type(16))) __bf16 v16bf;
typedef __attribute__((ext_vector_type(8)))  float  v8f;

union FragU {
  uint4 u[2];
  v16bf v;
};

static __device__ inline v16bf load_frag(const unsigned short* p0,
                                         const unsigned short* p1) {
  FragU f;
  f.u[0] = *reinterpret_cast<const uint4*>(p0);
  f.u[1] = *reinterpret_cast<const uint4*>(p1);
  return f.v;
}

static __device__ inline v8f v8f_zero() {
  v8f z;
#pragma unroll
  for (int i = 0; i < 8; ++i) z[i] = 0.0f;
  return z;
}

// round-to-nearest-even f32 -> bf16 (raw bits)
static __device__ inline unsigned short f2bf(float x) {
  unsigned int u = __float_as_uint(x);
  u += 0x7FFFu + ((u >> 16) & 1u);
  return (unsigned short)(u >> 16);
}

// 16B async copy global -> LDS (per-lane addresses), tracked by ASYNCcnt
static __device__ inline void async_copy16(unsigned lds_off,
                                           const unsigned short* g) {
  unsigned long long ga = (unsigned long long)(const void*)g;
  asm volatile("global_load_async_to_lds_b128 %0, %1, off"
               :: "v"(lds_off), "v"(ga)
               : "memory");
}

// ---------------- stage 1: f32 -> bf16 convert ----------------
__global__ void cvt_f32_to_bf16(const float* __restrict__ in,
                                unsigned short* __restrict__ out, int n) {
  int i = blockIdx.x * blockDim.x + threadIdx.x;
  int stride = gridDim.x * blockDim.x;
  for (; i < n; i += stride) out[i] = f2bf(in[i]);
}

// convert + transpose: in[Kdim][Ndim] f32 -> out[Ndim][Kdim] bf16
// (coalesced writes, strided reads; one-time cost on the weights)
__global__ void cvt_transpose_f32_to_bf16(const float* __restrict__ in,
                                          unsigned short* __restrict__ out,
                                          int Kdim, int Ndim) {
  int total = Kdim * Ndim;
  int i = blockIdx.x * blockDim.x + threadIdx.x;
  int stride = gridDim.x * blockDim.x;
  for (; i < total; i += stride) {
    int n = i / Kdim;
    int k = i - n * Kdim;
    out[i] = f2bf(in[(size_t)k * Ndim + n]);
  }
}

// ---------------- stage 2/4: tiled WMMA GEMM ----------------
// C[M,N] = A[M,K] * B[K,N] with B pre-transposed: Bt[N][K]. bf16 in, f32 accum.
// 128x128 block tile, 8 waves of 32x64, BK=32, double-buffered async staging.
// MODE 0: QKV epilogue -> Q(scaled)/K as [b,h,t,d], V transposed as [b,h,d,t]
// MODE 1: plain f32 row-major output
template <int MODE>
__global__ __launch_bounds__(256) void gemm_wmma_bf16(
    const unsigned short* __restrict__ A,   // [M][K]
    const unsigned short* __restrict__ Bt,  // [N][K]
    int M, int N, int K,
    unsigned short* __restrict__ outQ, unsigned short* __restrict__ outK,
    unsigned short* __restrict__ outVt, float* __restrict__ outF) {
  constexpr int BM = 128, BN = 128, BK = 32;
  __shared__ unsigned short As[2][BM * BK];  // [128][32] (M x K)
  __shared__ unsigned short Bs[2][BN * BK];  // [128][32] (N x K)

  const int tid  = threadIdx.x;
  const int wave = tid >> 5;
  const int lane = tid & 31;
  const int ln   = lane & 15;
  const int hi   = lane >> 4;
  const int wm   = wave >> 1;  // 0..3  (M direction)
  const int wn   = wave & 1;   // 0..1  (N direction)
  const int blockN = blockIdx.x * BN;
  const int blockM = blockIdx.y * BM;

  // stage one 128x32 tile pair via async global->LDS (4 x b128 per thread)
  auto stage = [&](int buf, int k0) {
    unsigned aB = (unsigned)(size_t)(void*)&As[buf][0];
    unsigned bB = (unsigned)(size_t)(void*)&Bs[buf][0];
#pragma unroll
    for (int s = 0; s < 2; ++s) {
      int e   = tid + s * 256;   // 0..511
      int row = e >> 2;          // 0..127
      int seg = (e & 3) * 8;     // 0,8,16,24
      async_copy16(aB + (unsigned)(row * BK + seg) * 2,
                   &A[(size_t)(blockM + row) * K + k0 + seg]);
      async_copy16(bB + (unsigned)(row * BK + seg) * 2,
                   &Bt[(size_t)(blockN + row) * K + k0 + seg]);
    }
  };

  v8f acc[2][4];
#pragma unroll
  for (int mt = 0; mt < 2; ++mt)
#pragma unroll
    for (int nt = 0; nt < 4; ++nt) acc[mt][nt] = v8f_zero();

  const int NK = K / BK;
  stage(0, 0);
  for (int i = 0; i < NK; ++i) {
    const int cur = i & 1;
    if (i + 1 < NK) {
      stage(1 - cur, (i + 1) * BK);
      // 4 newer async ops may remain in flight; tile i's ops complete in order
      asm volatile("s_wait_asynccnt 0x4" ::: "memory");
    } else {
      asm volatile("s_wait_asynccnt 0x0" ::: "memory");
    }
    __syncthreads();

    v16bf af[2], bf[4];
#pragma unroll
    for (int mt = 0; mt < 2; ++mt) {
      int row = wm * 32 + mt * 16 + ln;
      // A 16x32 bf16 layout: chunks at K = hi*8 and 16+hi*8
      af[mt] = load_frag(&As[cur][row * BK + hi * 8],
                         &As[cur][row * BK + 16 + hi * 8]);
    }
#pragma unroll
    for (int nt = 0; nt < 4; ++nt) {
      int col = wn * 64 + nt * 16 + ln;
      // B 32x16 bf16 layout: lane = column, chunks at K = hi*8 and 16+hi*8
      bf[nt] = load_frag(&Bs[cur][col * BK + hi * 8],
                         &Bs[cur][col * BK + 16 + hi * 8]);
    }
#pragma unroll
    for (int mt = 0; mt < 2; ++mt)
#pragma unroll
      for (int nt = 0; nt < 4; ++nt)
        acc[mt][nt] = __builtin_amdgcn_wmma_f32_16x16x32_bf16(
            false, af[mt], false, bf[nt], (short)0, acc[mt][nt], false, false);
    __syncthreads();  // everyone done reading buf 'cur' before it is re-staged
  }

  // --- epilogue ---
#pragma unroll
  for (int mt = 0; mt < 2; ++mt)
#pragma unroll
    for (int nt = 0; nt < 4; ++nt)
#pragma unroll
      for (int r = 0; r < 8; ++r) {
        int gm = blockM + wm * 32 + mt * 16 + r + hi * 8;
        int gn = blockN + wn * 64 + nt * 16 + ln;
        float val = acc[mt][nt][r];
        if (MODE == 0) {
          int b = gm / SEQ;
          int t = gm % SEQ;
          if (gn < DM) {                       // Q, pre-scaled by 1/sqrt(dk)
            int h = gn >> 6, d = gn & 63;
            outQ[(((size_t)(b * NH + h)) * SEQ + t) * HD + d] = f2bf(val * 0.125f);
          } else if (gn < 2 * DM) {            // K  [b,h,t,d]
            int c = gn - DM;
            int h = c >> 6, d = c & 63;
            outK[(((size_t)(b * NH + h)) * SEQ + t) * HD + d] = f2bf(val);
          } else {                             // V transposed [b,h,d,t]
            int c = gn - 2 * DM;
            int h = c >> 6, d = c & 63;
            outVt[(((size_t)(b * NH + h)) * HD + d) * SEQ + t] = f2bf(val);
          }
        } else {
          outF[(size_t)gm * N + gn] = val;
        }
      }
}

// ---------------- stage 3: causal flash attention ----------------
// One wave per 16-query tile. 32-key steps. Q pre-scaled.
__global__ __launch_bounds__(128) void flash_attn_wmma(
    const unsigned short* __restrict__ q,   // [b,h,t,d] bf16 (scaled)
    const unsigned short* __restrict__ k,   // [b,h,t,d] bf16
    const unsigned short* __restrict__ vt,  // [b,h,d,t] bf16
    unsigned short* __restrict__ attnOut) { // [b,t, h*64+d] bf16
  __shared__ unsigned short Plds[4][16 * 32];

  const int wave = threadIdx.x >> 5;
  const int lane = threadIdx.x & 31;
  const int ln = lane & 15;
  const int hi = lane >> 4;

  const int g  = blockIdx.x * 4 + wave;
  const int qt = g % (SEQ / 16);
  const int bh = g / (SEQ / 16);
  const int qBase = qt * 16;
  const int b = bh / NH, h = bh % NH;

  const unsigned short* qh  = q  + (size_t)bh * SEQ * HD;
  const unsigned short* kh  = k  + (size_t)bh * SEQ * HD;
  const unsigned short* vth = vt + (size_t)bh * HD * SEQ;

  // Q fragments for head-dim chunks [0,32) and [32,64); resident all loop.
  v16bf qa[2];
#pragma unroll
  for (int c = 0; c < 2; ++c) {
    const unsigned short* p = qh + (size_t)(qBase + ln) * HD + c * 32 + hi * 8;
    qa[c] = load_frag(p, p + 16);
  }

  v8f o[4];
#pragma unroll
  for (int n = 0; n < 4; ++n) o[n] = v8f_zero();
  float m[8], l[8];
#pragma unroll
  for (int r = 0; r < 8; ++r) { m[r] = -1e30f; l[r] = 0.0f; }

  unsigned short* P = Plds[wave];

  for (int kb = 0; kb < qBase + 16; kb += 32) {
    // prefetch next key tile (global_prefetch_b8)
    if (kb + 32 < qBase + 16) {
      __builtin_prefetch(kh + (size_t)(kb + 32 + ln) * HD, 0, 1);
      __builtin_prefetch(vth + (size_t)(16 + ln) * SEQ + kb + 32, 0, 1);
    }

    // S = Q * K^T for 2 key tiles of 16 (K matrix is B operand: contiguous d run)
    v8f s[2];
#pragma unroll
    for (int j = 0; j < 2; ++j) {
      const unsigned short* kp = kh + (size_t)(kb + j * 16 + ln) * HD + hi * 8;
      v16bf b0 = load_frag(kp, kp + 16);           // d in [0,32)
      v16bf b1 = load_frag(kp + 32, kp + 48);      // d in [32,64)
      v8f z = v8f_zero();
      z = __builtin_amdgcn_wmma_f32_16x16x32_bf16(false, qa[0], false, b0,
                                                  (short)0, z, false, false);
      s[j] = __builtin_amdgcn_wmma_f32_16x16x32_bf16(false, qa[1], false, b1,
                                                     (short)0, z, false, false);
    }

    const bool diag = (kb + 16 > qBase);  // tile touches the causal boundary

#pragma unroll
    for (int r = 0; r < 8; ++r) {
      int row = qBase + r + hi * 8;       // this lane's query row (C layout)
      float s0 = s[0][r], s1 = s[1][r];
      if (diag) {
        if (kb + ln > row)       s0 = -1e30f;
        if (kb + 16 + ln > row)  s1 = -1e30f;
      }
      // row max over the 16 lanes holding this row
      float mt = fmaxf(s0, s1);
#pragma unroll
      for (int off = 1; off < 16; off <<= 1)
        mt = fmaxf(mt, __shfl_xor(mt, off, 32));
      float mnew  = fmaxf(m[r], mt);
      float alpha = __expf(m[r] - mnew);
      float p0 = __expf(s0 - mnew);
      float p1 = __expf(s1 - mnew);
      float rs = p0 + p1;
#pragma unroll
      for (int off = 1; off < 16; off <<= 1)
        rs += __shfl_xor(rs, off, 32);
      l[r] = l[r] * alpha + rs;
      m[r] = mnew;
#pragma unroll
      for (int n = 0; n < 4; ++n) o[n][r] *= alpha;
      // write P (C layout -> LDS row-major 16x32)
      P[(r + hi * 8) * 32 + ln]      = f2bf(p0);
      P[(r + hi * 8) * 32 + 16 + ln] = f2bf(p1);
    }

    // make cross-lane LDS writes visible before A-layout reload (wave-private)
    asm volatile("s_wait_dscnt 0" ::: "memory");

    v16bf pa = load_frag(&P[ln * 32 + hi * 8], &P[ln * 32 + 16 + hi * 8]);
#pragma unroll
    for (int n = 0; n < 4; ++n) {
      const unsigned short* vp = vth + (size_t)(n * 16 + ln) * SEQ + kb + hi * 8;
      v16bf vb = load_frag(vp, vp + 16);
      o[n] = __builtin_amdgcn_wmma_f32_16x16x32_bf16(false, pa, false, vb,
                                                     (short)0, o[n], false, false);
    }
  }

  // epilogue: normalize and write bf16 attention output [b,t,D]
#pragma unroll
  for (int r = 0; r < 8; ++r) {
    float inv = 1.0f / l[r];
    int row = qBase + r + hi * 8;
    size_t base = ((size_t)b * SEQ + row) * DM + h * HD;
#pragma unroll
    for (int n = 0; n < 4; ++n)
      attnOut[base + n * 16 + ln] = f2bf(o[n][r] * inv);
  }
}

// ---------------- launch ----------------
extern "C" void kernel_launch(void* const* d_in, const int* in_sizes, int n_in,
                              void* d_out, int out_size, void* d_ws, size_t ws_size,
                              hipStream_t stream) {
  const float* x      = (const float*)d_in[0];
  // d_in[1] = attn_mask : causal mask is applied analytically, input unused
  const float* w_qkv  = (const float*)d_in[2];
  const float* w_proj = (const float*)d_in[3];
  float* out = (float*)d_out;

  char* ws = (char*)d_ws;
  size_t off = 0;
  unsigned short* xb      = (unsigned short*)(ws + off); off += (size_t)M_ROWS * DM * 2;  // 16 MB
  unsigned short* wqkvT   = (unsigned short*)(ws + off); off += (size_t)DM * 3 * DM * 2;  //  6 MB
  unsigned short* wprojT  = (unsigned short*)(ws + off); off += (size_t)DM * DM * 2;      //  2 MB
  unsigned short* qs      = (unsigned short*)(ws + off); off += (size_t)M_ROWS * DM * 2;  // 16 MB
  unsigned short* ks      = (unsigned short*)(ws + off); off += (size_t)M_ROWS * DM * 2;  // 16 MB
  unsigned short* vts     = (unsigned short*)(ws + off); off += (size_t)M_ROWS * DM * 2;  // 16 MB
  unsigned short* attnb   = (unsigned short*)(ws + off); off += (size_t)M_ROWS * DM * 2;  // 16 MB

  // stage 1: bf16 conversion (weights transposed to [N][K] for the GEMMs)
  cvt_f32_to_bf16<<<2048, 256, 0, stream>>>(x, xb, M_ROWS * DM);
  cvt_transpose_f32_to_bf16<<<1024, 256, 0, stream>>>(w_qkv, wqkvT, DM, 3 * DM);
  cvt_transpose_f32_to_bf16<<<512, 256, 0, stream>>>(w_proj, wprojT, DM, DM);

  // stage 2: QKV projection (scatter epilogue into flash layouts)
  dim3 g1(3 * DM / 128, M_ROWS / 128);
  gemm_wmma_bf16<0><<<g1, 256, 0, stream>>>(xb, wqkvT, M_ROWS, 3 * DM, DM,
                                            qs, ks, vts, nullptr);

  // stage 3: causal flash attention (1 wave per 16-query tile)
  int qtiles = BATCH * NH * (SEQ / 16);  // 8192
  flash_attn_wmma<<<qtiles / 4, 128, 0, stream>>>(qs, ks, vts, attnb);

  // stage 4: output projection -> f32
  dim3 g2(DM / 128, M_ROWS / 128);
  gemm_wmma_bf16<1><<<g2, 256, 0, stream>>>(attnb, wprojT, M_ROWS, DM, DM,
                                            nullptr, nullptr, nullptr, out);
}
